// MultiHeadAttention_71502615544564
// MI455X (gfx1250) — compile-verified, matrix operand order checked by reading, not run on hardware
//
#include <hip/hip_runtime.h>
#include <hip/hip_bf16.h>

// ---------------------------------------------------------------------------
// Problem constants (from the reference)
// ---------------------------------------------------------------------------
#define NNODES   50000      // = 3125 * 16, exact multiple of WMMA M tile
#define DIM      128        // K for both GEMMs
#define HEADS    8
#define HEAD_DIM 16
#define NEDGES   640000
#define QKV_COLS 384        // per head h: q=cols[h*48+0..15], k=+16, v=+32
#define MTILES   (NNODES / 16)   // 3125

typedef __attribute__((ext_vector_type(16))) __bf16 v16bf;
typedef __attribute__((ext_vector_type(8)))  __bf16 v8bf;
typedef __attribute__((ext_vector_type(8)))  float  v8f;

// ---------------------------------------------------------------------------
// TDM availability / arity detection
// ---------------------------------------------------------------------------
#ifndef __has_builtin
#define __has_builtin(x) 0
#endif
#if __has_builtin(__builtin_amdgcn_tensor_load_to_lds)
#define HAVE_TDM 1
#if defined(__has_include)
#if __has_include(<hip/amd_detail/amd_gfx1250_TDM.h>)
#define TDM_SIX_ARG 1   // therock-10.0 headers -> 6-arg builtin
#endif
#endif
#endif
#if __has_builtin(__builtin_amdgcn_s_wait_tensorcnt)
#define HAVE_TENSORCNT 1
#endif

typedef __attribute__((ext_vector_type(4))) unsigned tdm_u32x4;
typedef __attribute__((ext_vector_type(8))) int      tdm_i32x8;
typedef __attribute__((ext_vector_type(4))) int      tdm_i32x4;

// Issue a TDM load of a 2D tile (tile_rows x 128 bf16 elements, row stride 128)
// from global `gaddr` into LDS byte-offset `lds_addr`. Call from ONE wave only.
__device__ __forceinline__ void tdm_load_rows_128bf16(unsigned lds_addr,
                                                      unsigned long long gaddr,
                                                      unsigned tile_rows,
                                                      unsigned tensor_rows) {
#if defined(HAVE_TDM)
    // D# group 0 (ISA 8.3): count=1 | lds_addr | global_addr | type=2
    tdm_u32x4 g0;
    g0[0] = 1u;                                       // count=1, user mode
    g0[1] = lds_addr;                                 // LDS byte address
    g0[2] = (unsigned)(gaddr & 0xFFFFFFFFu);          // global_addr[31:0]
    g0[3] = (unsigned)((gaddr >> 32) & 0x1FFFFFFu)    // global_addr[56:32]
            | (2u << 30);                             // type=2 ("image")
    // D# group 1 (ISA 8.4)
    tdm_i32x8 g1;
    g1[0] = (int)(1u << 16);                          // data_size=1 (2 bytes)
    g1[1] = (int)((128u & 0xFFFFu) << 16);            // tensor_dim0=128 (bits 79:48)
    g1[2] = (int)((tensor_rows & 0xFFFFu) << 16);     // dim0 hi=0 | tensor_dim1 lo
    g1[3] = (int)((tensor_rows >> 16) & 0xFFFFu)      // tensor_dim1 hi
            | (int)(128u << 16);                      // tile_dim0=128 (bits 127:112)
    g1[4] = (int)(tile_rows & 0xFFFFu);               // tile_dim1, tile_dim2=0
    g1[5] = 128;                                      // tensor_dim0_stride = 128
    g1[6] = 0;
    g1[7] = 0;
    tdm_i32x4 g2 = {0, 0, 0, 0};
    tdm_i32x4 g3 = {0, 0, 0, 0};
#if defined(TDM_SIX_ARG)
    tdm_i32x8 g4 = {0, 0, 0, 0, 0, 0, 0, 0};
    __builtin_amdgcn_tensor_load_to_lds(g0, g1, g2, g3, g4, 0);
#else
    __builtin_amdgcn_tensor_load_to_lds(g0, g1, g2, g3, 0);
#endif
#if defined(HAVE_TENSORCNT)
    __builtin_amdgcn_s_wait_tensorcnt(0);
#else
    asm volatile("s_wait_tensorcnt 0x0" ::: "memory");
#endif
#else
    (void)lds_addr; (void)gaddr; (void)tile_rows; (void)tensor_rows;
#endif
}

// ---------------------------------------------------------------------------
// Kernel 0: zero-fill
// ---------------------------------------------------------------------------
__global__ __launch_bounds__(256) void zero_f32(float* __restrict__ p, long n) {
    long i = (long)blockIdx.x * blockDim.x + threadIdx.x;
    if (i < n) p[i] = 0.0f;
}

// ---------------------------------------------------------------------------
// Prepass: fp32 -> bf16 row-major copy (for x / accum A-operands)
// ---------------------------------------------------------------------------
__global__ __launch_bounds__(256)
void cvt_bf16(const float* __restrict__ in, __bf16* __restrict__ out, long n) {
    long i = ((long)blockIdx.x * blockDim.x + threadIdx.x) * 4;
    if (i + 3 < n) {
        const float4 v = *(const float4*)(in + i);
        out[i + 0] = (__bf16)v.x;
        out[i + 1] = (__bf16)v.y;
        out[i + 2] = (__bf16)v.z;
        out[i + 3] = (__bf16)v.w;
    }
}

// ---------------------------------------------------------------------------
// Prepass: W[K x NC] fp32 -> Wt[NC x K] bf16 (column-major => contiguous K runs)
// ---------------------------------------------------------------------------
__global__ __launch_bounds__(256)
void transpose_w_bf16(const float* __restrict__ W, __bf16* __restrict__ Wt,
                      int K, int NC) {
    long i = (long)blockIdx.x * blockDim.x + threadIdx.x;
    if (i >= (long)K * NC) return;
    int k = (int)(i / NC), c = (int)(i % NC);
    Wt[(long)c * K + k] = (__bf16)W[i];
}

// ---------------------------------------------------------------------------
// Fragment loaders (two contiguous 16B runs per lane, per ISA 7.12.2 striping)
// ---------------------------------------------------------------------------
__device__ __forceinline__ v16bf load_frag(const __bf16* p) {
    const v8bf lo = *(const v8bf*)(p);
    const v8bf hi = *(const v8bf*)(p + 16);
    v16bf f;
    #pragma unroll
    for (int i = 0; i < 8; ++i) { f[i] = lo[i]; f[8 + i] = hi[i]; }
    return f;
}

// ---------------------------------------------------------------------------
// GEMM: C[M x NC] = A[M x 128] * W[128 x NC] + bias  (bf16 WMMA, f32 accum)
//   A   : bf16 row-major [M x 128]
//   Wt  : bf16 column-major [NC x 128] (row c = column c of W)
//   Block: 256 threads = 8 waves; grid (ceil(3125/8), NC/64).
//   The block's 64-column slab of Wt (64x128 bf16 = 16 KB) is DMA'd into LDS
//   by the Tensor Data Mover; each wave computes one 16x64 strip.
//   Per K-chunk: 1 A fragment (2x global b128), then ALL four B fragments
//   (8x ds_load_b128, one clause/one wait), then 4 back-to-back WMMAs.
//
// Fragment layouts per CDNA5 ISA 7.12.2 (wave32):
//   A 16x32 bf16 : lane {half=lane>>4, m=lane&15}; elem i holds
//                  K = (i<8 ? half*8+i : 16+half*8+(i-8)) of row m.
//   B 32x16 bf16 : mirrored, lane&15 = output column, same K striping.
//   C/D f32      : acc[r] = C[M = r + 8*half, N = lane&15].
// ---------------------------------------------------------------------------
template<int NC>
__global__ __launch_bounds__(256)
void wmma_gemm_bias(const __bf16* __restrict__ A, const __bf16* __restrict__ Wt,
                    const float* __restrict__ bias, float* __restrict__ C) {
    __shared__ __bf16 wtile[64 * DIM];     // 16 KB: 64 columns x 128 K

    const int group = blockIdx.y;          // which 64-column slab
    const unsigned long long wt_g =
        (unsigned long long)(const void*)(Wt + (size_t)group * 64 * DIM);

#if defined(HAVE_TDM)
    if (threadIdx.x == 0) {
        tdm_load_rows_128bf16((unsigned)(unsigned long long)(void*)wtile,
                              wt_g, 64u, 64u);
    }
#else
    {   // cooperative fallback copy: 16 KB = 1024 x 16B
        const uint4* src = (const uint4*)(Wt + (size_t)group * 64 * DIM);
        uint4* dst = (uint4*)wtile;
        for (int i = threadIdx.x; i < 64 * DIM * 2 / 16; i += blockDim.x)
            dst[i] = src[i];
    }
#endif
    __syncthreads();

    const int lane  = threadIdx.x & 31;
    const int wave  = threadIdx.x >> 5;
    const int mtile = blockIdx.x * 8 + wave;
    if (mtile >= MTILES) return;

    const int  half = lane >> 4;
    const int  l16  = lane & 15;
    const long row  = (long)mtile * 16 + l16;

    v8f acc0 = {}, acc1 = {}, acc2 = {}, acc3 = {};
    #pragma unroll
    for (int kb = 0; kb < DIM; kb += 32) {
        const int koff = kb + half * 8;
        // A fragment (global, 2x b128)
        const v16bf a = load_frag(A + row * DIM + koff);
        // All four B fragments up front (8x ds_load_b128 -> single wait)
        const v16bf b0 = load_frag(wtile + (0 * 16 + l16) * DIM + koff);
        const v16bf b1 = load_frag(wtile + (1 * 16 + l16) * DIM + koff);
        const v16bf b2 = load_frag(wtile + (2 * 16 + l16) * DIM + koff);
        const v16bf b3 = load_frag(wtile + (3 * 16 + l16) * DIM + koff);
        // 4 back-to-back WMMAs, independent accumulators, shared A
        acc0 = __builtin_amdgcn_wmma_f32_16x16x32_bf16(
            false, a, false, b0, (short)0, acc0, false, false);
        acc1 = __builtin_amdgcn_wmma_f32_16x16x32_bf16(
            false, a, false, b1, (short)0, acc1, false, false);
        acc2 = __builtin_amdgcn_wmma_f32_16x16x32_bf16(
            false, a, false, b2, (short)0, acc2, false, false);
        acc3 = __builtin_amdgcn_wmma_f32_16x16x32_bf16(
            false, a, false, b3, (short)0, acc3, false, false);
    }

    #pragma unroll
    for (int sub = 0; sub < 4; ++sub) {
        const int col = group * 64 + sub * 16 + l16;
        const float bv = bias[col];
        const v8f acc = (sub == 0) ? acc0 : (sub == 1) ? acc1
                                          : (sub == 2) ? acc2 : acc3;
        #pragma unroll
        for (int r = 0; r < 8; ++r) {
            long orow = (long)mtile * 16 + r + half * 8;
            C[orow * NC + col] = acc[r] + bv;
        }
    }
}

// ---------------------------------------------------------------------------
// Monotone uint key for float atomic-max (key order == float order).
// ---------------------------------------------------------------------------
__device__ __forceinline__ unsigned f32_key(float f) {
    unsigned b = __float_as_uint(f);
    return b ^ (unsigned)(((int)b >> 31) | 0x80000000);
}
__device__ __forceinline__ float key_f32(unsigned k) {
    unsigned b = (k & 0x80000000u) ? (k ^ 0x80000000u) : ~k;
    return __uint_as_float(b);
}

// ---------------------------------------------------------------------------
// Kernel 2: per-edge, per-head attention scores + global per-head max.
// ---------------------------------------------------------------------------
__global__ __launch_bounds__(256)
void edge_scores(const float* __restrict__ qkv, const long long* __restrict__ ei,
                 float* __restrict__ attn, unsigned* __restrict__ maxkey) {
    __shared__ unsigned smax[HEADS];
    if (threadIdx.x < HEADS) smax[threadIdx.x] = 0u;   // key 0 <= key(-inf)
    __syncthreads();

    const int  lane = threadIdx.x & 31;
    const int  wave = threadIdx.x >> 5;
    const long e    = (long)blockIdx.x * 8 + wave;
    if (e < NEDGES) {
        const int  h  = lane >> 2;
        const int  d0 = (lane & 3) * 4;
        const long s  = ei[e];
        const long d  = ei[NEDGES + e];
        const float4 q = *(const float4*)(qkv + s * QKV_COLS + h * 48 + d0);
        const float4 k = *(const float4*)(qkv + d * QKV_COLS + h * 48 + 16 + d0);
        float p = q.x * k.x + q.y * k.y + q.z * k.z + q.w * k.w;
        p += __shfl_xor(p, 1, 32);
        p += __shfl_xor(p, 2, 32);
        if ((lane & 3) == 0) {
            const float score = p * 0.25f;  // 1/sqrt(HEAD_DIM)
            attn[e * HEADS + h] = score;
            atomicMax(&smax[h], f32_key(score));
        }
    }
    __syncthreads();
    if (threadIdx.x < HEADS) atomicMax(&maxkey[threadIdx.x], smax[threadIdx.x]);
}

// ---------------------------------------------------------------------------
// Kernel 3: exp(attn - max[h]) in place, segment-sum by dst.
// ---------------------------------------------------------------------------
__global__ __launch_bounds__(256)
void exp_segsum(const long long* __restrict__ ei, float* __restrict__ attn,
                const unsigned* __restrict__ maxkey, float* __restrict__ sum_exp) {
    const long idx = (long)blockIdx.x * blockDim.x + threadIdx.x;
    if (idx >= (long)NEDGES * HEADS) return;
    const long e = idx >> 3;
    const int  h = (int)(idx & 7);
    const float mx = key_f32(maxkey[h]);
    const float w  = __expf(attn[idx] - mx);
    attn[idx] = w;
    const long d = ei[NEDGES + e];
    atomicAdd(&sum_exp[d * HEADS + h], w);
}

// ---------------------------------------------------------------------------
// Kernel 4: accum[dst] += v[dst] * (exp / (sum_exp[dst] + 1e-8)).
// ---------------------------------------------------------------------------
__global__ __launch_bounds__(256)
void aggregate(const float* __restrict__ qkv, const long long* __restrict__ ei,
               const float* __restrict__ expw, const float* __restrict__ sum_exp,
               float* __restrict__ accum) {
    const int  lane = threadIdx.x & 31;
    const int  wave = threadIdx.x >> 5;
    const long e    = (long)blockIdx.x * 8 + wave;
    if (e >= NEDGES) return;
    const int  h  = lane >> 2;
    const int  d0 = (lane & 3) * 4;
    const long d  = ei[NEDGES + e];
    const float w = expw[e * HEADS + h] / (sum_exp[d * HEADS + h] + 1e-8f);
    const float4 v = *(const float4*)(qkv + d * QKV_COLS + h * 48 + 32 + d0);
    float* o = accum + d * DIM + h * HEAD_DIM + d0;
    atomicAdd(o + 0, v.x * w);
    atomicAdd(o + 1, v.y * w);
    atomicAdd(o + 2, v.z * w);
    atomicAdd(o + 3, v.w * w);
}

// ---------------------------------------------------------------------------
// Launch
// ---------------------------------------------------------------------------
extern "C" void kernel_launch(void* const* d_in, const int* in_sizes, int n_in,
                              void* d_out, int out_size, void* d_ws, size_t ws_size,
                              hipStream_t stream) {
    (void)in_sizes; (void)n_in; (void)out_size; (void)ws_size;

    const float*     x    = (const float*)d_in[0];
    const long long* ei   = (const long long*)d_in[1];  // int64 edge_index (2,E)
    const float*     Wqkv = (const float*)d_in[2];
    const float*     bqkv = (const float*)d_in[3];
    const float*     Wout = (const float*)d_in[4];
    const float*     bout = (const float*)d_in[5];
    float*           out  = (float*)d_out;

    char* ws = (char*)d_ws;
    size_t off = 0;
    auto wsalloc = [&](size_t bytes) -> void* {
        void* p = ws + off;
        off = (off + bytes + 255) & ~(size_t)255;
        return p;
    };
    float*    qkv     = (float*)wsalloc((size_t)NNODES * QKV_COLS * 4);  // 76.8 MB
    float*    attn    = (float*)wsalloc((size_t)NEDGES * HEADS * 4);     // 20.5 MB
    float*    sum_exp = (float*)wsalloc((size_t)NNODES * HEADS * 4);     //  1.6 MB
    float*    accum   = (float*)wsalloc((size_t)NNODES * DIM * 4);       // 25.6 MB
    __bf16*   xbf     = (__bf16*)wsalloc((size_t)NNODES * DIM * 2);      // 12.8 MB
    __bf16*   abf     = (__bf16*)wsalloc((size_t)NNODES * DIM * 2);      // 12.8 MB
    __bf16*   WqkvT   = (__bf16*)wsalloc((size_t)QKV_COLS * DIM * 2);    //   98 KB
    __bf16*   WoutT   = (__bf16*)wsalloc((size_t)DIM * DIM * 2);         //   32 KB
    unsigned* maxkey  = (unsigned*)wsalloc(HEADS * 4);

    // Zero atomically-accumulated regions (key 0 == -inf sentinel for maxkey).
    {
        long n = (long)NNODES * HEADS;
        zero_f32<<<dim3((unsigned)((n + 255) / 256)), 256, 0, stream>>>(sum_exp, n);
        n = (long)NNODES * DIM;
        zero_f32<<<dim3((unsigned)((n + 255) / 256)), 256, 0, stream>>>(accum, n);
        zero_f32<<<1, 256, 0, stream>>>((float*)maxkey, HEADS);
    }

    // Prepass: bf16 operands for the WMMA GEMMs.
    {
        long n = (long)NNODES * DIM;                       // x -> bf16
        cvt_bf16<<<dim3((unsigned)(n / 4 / 256)), 256, 0, stream>>>(x, xbf, n);
        transpose_w_bf16<<<dim3((128 * QKV_COLS + 255) / 256), 256, 0, stream>>>(
            Wqkv, WqkvT, DIM, QKV_COLS);
        transpose_w_bf16<<<dim3((128 * DIM + 255) / 256), 256, 0, stream>>>(
            Wout, WoutT, DIM, DIM);
    }

    const unsigned gemm_gx = (MTILES + 7) / 8;   // 391

    // 1) qkv = x @ W_qkv + b_qkv   (TDM-staged LDS weights + bf16 WMMA)
    wmma_gemm_bias<QKV_COLS><<<dim3(gemm_gx, QKV_COLS / 64), 256, 0, stream>>>(
        xbf, WqkvT, bqkv, qkv);

    // 2) per-edge scores + global per-head max
    edge_scores<<<dim3(NEDGES / 8), 256, 0, stream>>>(qkv, ei, attn, maxkey);

    // 3) exp + segment sum by dst
    {
        long n = (long)NEDGES * HEADS;
        exp_segsum<<<dim3((unsigned)((n + 255) / 256)), 256, 0, stream>>>(
            ei, attn, maxkey, sum_exp);
    }

    // 4) weighted scatter of v into accum
    aggregate<<<dim3(NEDGES / 8), 256, 0, stream>>>(qkv, ei, attn, sum_exp, accum);

    // 5) out = accum @ W_out + b_out
    {
        long n = (long)NNODES * DIM;                       // accum -> bf16
        cvt_bf16<<<dim3((unsigned)(n / 4 / 256)), 256, 0, stream>>>(accum, abf, n);
        wmma_gemm_bias<DIM><<<dim3(gemm_gx, DIM / 64), 256, 0, stream>>>(
            abf, WoutT, bout, out);
    }
}
